// IDCT2_1133871366829
// MI455X (gfx1250) — matrix operand order repaired and checked by reading, not compiled, not run on hardware
//
#include <hip/hip_runtime.h>
#include <math.h>

// ---------------------------------------------------------------------------
// 2-D IDCT (DCT-III both axes) of 4096x4096 fp32 as Y = M * X * M^T.
// Compute-bound (2.75e11 FLOP vs ~256MB HBM) -> use the fast matrix path:
// V_WMMA_F32_16X16X32_F16 with split-fp16 (hi+lo) operands, fp32 accumulate:
//   x*y ~= xh*yh + xh*yl + xl*yh   (3 f16 WMMAs ~ 22-bit mantissa accuracy)
// Two passes of C = (M*A)^T:  pass1 W1 = (M*X)^T,  pass2 Y = (M*W1)^T.
// Workspace: Mh (32MB) + Ml (32MB) + W1 f32 (64MB) = 128MB.
// ---------------------------------------------------------------------------

typedef float    v8f  __attribute__((ext_vector_type(8)));
typedef _Float16 v8h  __attribute__((ext_vector_type(8)));
typedef _Float16 v16h __attribute__((ext_vector_type(16)));

constexpr int NN   = 4096;
constexpr int BM   = 128;      // C-tile rows (rows of P = M*A)
constexpr int BN   = 128;      // C-tile cols
constexpr int KT   = 32;       // K panel == one WMMA K
constexpr int LDAH = 40;       // halves per A-tile row: 80B = 20 banks ->
                               // 16 lanes' 4-bank windows partition 0..63
constexpr int LDBH = 40;       // halves per B-tile *column* (column-major tile)

// --- Kernel 1: DCT-III basis, pre-split into f16 hi/lo ---------------------
// M[r][q] = 0.5 if q==0 else cos(pi*q*(2r+1)/(2N)); exact reduction mod 4N.
__global__ __launch_bounds__(256) void idct_fill_basis_f16(
    _Float16* __restrict__ Mh, _Float16* __restrict__ Ml)
{
    int idx = blockIdx.x * 256 + threadIdx.x;
    int r = idx >> 12;
    int q = idx & (NN - 1);
    float val;
    if (q == 0) {
        val = 0.5f;
    } else {
        unsigned m = ((unsigned)q * (unsigned)(2 * r + 1)) & 16383u;
        val = cospif((float)m * (1.0f / 8192.0f));
    }
    _Float16 h = (_Float16)val;
    Mh[idx] = h;
    Ml[idx] = (_Float16)(val - (float)h);
}

// --- Kernel 2: C = (M * A)^T via split-f16 WMMA ----------------------------
__global__ __launch_bounds__(256) void idct_gemmT_f16(
    float* __restrict__ C,
    const _Float16* __restrict__ Mh, const _Float16* __restrict__ Ml,
    const float* __restrict__ A)
{
    __shared__ _Float16 Ash[2][BM * LDAH];   // 10KB each buffer
    __shared__ _Float16 Asl[2][BM * LDAH];
    __shared__ _Float16 Bsh[2][BN * LDBH];   // column-major activation tiles
    __shared__ _Float16 Bsl[2][BN * LDBH];   // total 80KB

    const int tid  = threadIdx.x;
    const int lane = tid & 31;
    const int wid  = tid >> 5;          // 8 waves
    const int wrow = wid >> 2;          // 0..1  -> 64-row wave tile
    const int wcol = wid & 3;           // 0..3  -> 32-col wave tile
    const int ln   = lane & 15;
    const int lh   = lane >> 4;

    const int rb = blockIdx.y * BM;     // P row block (rows of M)
    const int cb = blockIdx.x * BN;     // P col block (cols of A)

    // A-side: 512 16B chunks per f16 array, 2 per thread (rows +0 / +64)
    const int a_row = tid >> 2;
    const int a_c   = (tid & 3) * 8;    // halves within the 32-half row
    // B-side: each thread owns one column and 16 consecutive K's
    const int b_col = tid & 127;
    const int b_kh  = tid >> 7;         // 0 or 1

    v8f acc[4][2];
#pragma unroll
    for (int mi = 0; mi < 4; ++mi)
#pragma unroll
        for (int ni = 0; ni < 2; ++ni)
            acc[mi][ni] = (v8f){0.f, 0.f, 0.f, 0.f, 0.f, 0.f, 0.f, 0.f};

    auto loadA = [&](int q0, uint4& h0, uint4& h1, uint4& l0, uint4& l1) {
        const _Float16* pH = Mh + (size_t)(rb + a_row) * NN + q0 + a_c;
        const _Float16* pL = Ml + (size_t)(rb + a_row) * NN + q0 + a_c;
        h0 = *(const uint4*)pH;
        h1 = *(const uint4*)(pH + (size_t)64 * NN);
        l0 = *(const uint4*)pL;
        l1 = *(const uint4*)(pL + (size_t)64 * NN);
    };
    auto loadB = [&](int q0, float* x) {          // 16 coalesced f32 loads
        const float* p = A + (size_t)(q0 + b_kh * 16) * NN + cb + b_col;
#pragma unroll
        for (int i = 0; i < 16; ++i) x[i] = p[(size_t)i * NN];
    };
    auto commit = [&](int buf, uint4 h0, uint4 h1, uint4 l0, uint4 l1,
                      const float* x) {
        *(uint4*)&Ash[buf][a_row * LDAH + a_c]        = h0;
        *(uint4*)&Ash[buf][(a_row + 64) * LDAH + a_c] = h1;
        *(uint4*)&Asl[buf][a_row * LDAH + a_c]        = l0;
        *(uint4*)&Asl[buf][(a_row + 64) * LDAH + a_c] = l1;
        v8h hh0, hh1, ll0, ll1;
#pragma unroll
        for (int i = 0; i < 8; ++i) {
            _Float16 h = (_Float16)x[i];
            hh0[i] = h;  ll0[i] = (_Float16)(x[i] - (float)h);
            _Float16 g = (_Float16)x[i + 8];
            hh1[i] = g;  ll1[i] = (_Float16)(x[i + 8] - (float)g);
        }
        const int ob = b_col * LDBH + b_kh * 16;
        *(v8h*)&Bsh[buf][ob]     = hh0;
        *(v8h*)&Bsh[buf][ob + 8] = hh1;
        *(v8h*)&Bsl[buf][ob]     = ll0;
        *(v8h*)&Bsl[buf][ob + 8] = ll1;
    };

    // ---- preload stage 0 ----
    {
        uint4 h0, h1, l0, l1; float x[16];
        loadA(0, h0, h1, l0, l1);
        loadB(0, x);
        commit(0, h0, h1, l0, l1, x);
    }
    __syncthreads();

    const int NK = NN / KT;             // 128 K stages
    for (int ks = 0; ks < NK; ++ks) {
        const int cur = ks & 1;

        uint4 h0, h1, l0, l1; float x[16];
        const bool more = (ks + 1 < NK);
        if (more) {                     // prefetch next stage into registers
            loadA((ks + 1) * KT, h0, h1, l0, l1);
            loadB((ks + 1) * KT, x);
        }

        // ---- B fragments (per-lane column K-runs, contiguous b128s) ----
        v16h bh[2], bl[2];
#pragma unroll
        for (int ni = 0; ni < 2; ++ni) {
            const int col = wcol * 32 + ni * 16 + ln;
            const int ob  = col * LDBH + 16 * lh;
            v8h p0 = *(const v8h*)&Bsh[cur][ob];
            v8h p1 = *(const v8h*)&Bsh[cur][ob + 8];
            bh[ni] = __builtin_shufflevector(p0, p1, 0, 1, 2, 3, 4, 5, 6, 7,
                                             8, 9, 10, 11, 12, 13, 14, 15);
            v8h q0v = *(const v8h*)&Bsl[cur][ob];
            v8h q1v = *(const v8h*)&Bsl[cur][ob + 8];
            bl[ni] = __builtin_shufflevector(q0v, q1v, 0, 1, 2, 3, 4, 5, 6, 7,
                                             8, 9, 10, 11, 12, 13, 14, 15);
        }

        // ---- A fragments + 3-term split-f16 WMMA ----
#pragma unroll
        for (int mi = 0; mi < 4; ++mi) {
            const int row = wrow * 64 + mi * 16 + ln;
            const int oa  = row * LDAH + 16 * lh;
            v8h a0 = *(const v8h*)&Ash[cur][oa];
            v8h a1 = *(const v8h*)&Ash[cur][oa + 8];
            v16h ah = __builtin_shufflevector(a0, a1, 0, 1, 2, 3, 4, 5, 6, 7,
                                              8, 9, 10, 11, 12, 13, 14, 15);
            v8h c0 = *(const v8h*)&Asl[cur][oa];
            v8h c1 = *(const v8h*)&Asl[cur][oa + 8];
            v16h al = __builtin_shufflevector(c0, c1, 0, 1, 2, 3, 4, 5, 6, 7,
                                              8, 9, 10, 11, 12, 13, 14, 15);
#pragma unroll
            for (int ni = 0; ni < 2; ++ni) {
                acc[mi][ni] = __builtin_amdgcn_wmma_f32_16x16x32_f16(
                    false, ah, false, bh[ni], (short)0, acc[mi][ni], false, false);
                acc[mi][ni] = __builtin_amdgcn_wmma_f32_16x16x32_f16(
                    false, ah, false, bl[ni], (short)0, acc[mi][ni], false, false);
                acc[mi][ni] = __builtin_amdgcn_wmma_f32_16x16x32_f16(
                    false, al, false, bh[ni], (short)0, acc[mi][ni], false, false);
            }
        }

        if (more) commit(cur ^ 1, h0, h1, l0, l1, x);
        __syncthreads();
    }

    // ---- transposed epilogue: C[col][row] = P[row][col] -------------------
    // Lane holds P[v + 8*lh][ln] (v=0..7): 8 contiguous floats of one C row.
    const int rbase = rb + wrow * 64;
    const int cbase = cb + wcol * 32;
#pragma unroll
    for (int mi = 0; mi < 4; ++mi) {
#pragma unroll
        for (int ni = 0; ni < 2; ++ni) {
            float* cp = &C[(size_t)(cbase + ni * 16 + ln) * NN +
                           rbase + mi * 16 + 8 * lh];
            v8f v = acc[mi][ni];
            float4 lo = {v[0], v[1], v[2], v[3]};
            float4 hi = {v[4], v[5], v[6], v[7]};
            *(float4*)(cp + 0) = lo;
            *(float4*)(cp + 4) = hi;
        }
    }
}

extern "C" void kernel_launch(void* const* d_in, const int* in_sizes, int n_in,
                              void* d_out, int out_size, void* d_ws, size_t ws_size,
                              hipStream_t stream)
{
    (void)in_sizes; (void)n_in; (void)out_size; (void)ws_size;
    const float* X   = (const float*)d_in[0];
    float*       out = (float*)d_out;
    _Float16*    Mh  = (_Float16*)d_ws;                 // 32 MB
    _Float16*    Ml  = Mh + (size_t)NN * NN;            // 32 MB
    float*       W1  = (float*)(Ml + (size_t)NN * NN);  // 64 MB intermediate

    idct_fill_basis_f16<<<(NN * NN) / 256, 256, 0, stream>>>(Mh, Ml);

    dim3 grid(NN / BN, NN / BM);       // 32 x 32 workgroups
    idct_gemmT_f16<<<grid, 256, 0, stream>>>(W1, Mh, Ml, X);   // W1 = (M X)^T
    idct_gemmT_f16<<<grid, 256, 0, stream>>>(out, Mh, Ml, W1); // Y = (M W1)^T
}
